// CINBrock_69509750718709
// MI455X (gfx1250) — matrix operand order repaired and checked by reading, not compiled, not run on hardware
//
#include <hip/hip_runtime.h>
#include <hip/hip_bf16.h>
#include <stdint.h>

// out[b,o] = sum_{f,p} x0[b,f]*x[b,p]*W[o,f,p] + bias[o]
// == GEMM C[4096,256] = A[4096,16384] x W^T with A[b, f*128+p] = x0[b,f]*x[b,p]
// A built on the fly from the outer-product structure. bf16 WMMA, f32 accumulate.

typedef __attribute__((ext_vector_type(16))) __bf16 v16bf;
typedef __attribute__((ext_vector_type(8)))  float  v8f;

#define B_TOT 4096
#define FIELD 128
#define HPREV 128
#define HNEXT 256
#define KTOT  (FIELD * HPREV)

#define BM   128   // b rows per workgroup
#define BN   64    // o cols per workgroup
#define FB   4     // f values staged per super-step
#define XROW 17    // uint4 per sX row (16 data = 128 bf16, +1 pad vs bank conflicts)
#define WROW 5     // uint4 per sW (ff,n) row (4 data = 32 bf16, +1 pad)

union Frag {
  v16bf    v;
  uint4    u2[2];
  unsigned w[8];
};
union F4 { float4 v; float f[4]; };

// one-instruction bf16 splat pair: v_cvt_pk_bf16_f32 d, f, f
__device__ __forceinline__ unsigned splat_bf16(float f) {
  unsigned d;
  asm("v_cvt_pk_bf16_f32 %0, %1, %1" : "=v"(d) : "v"(f));
  return d;
}

// packed bf16 multiply (V_PK_MUL_BF16)
__device__ __forceinline__ unsigned pk_mul_bf16(unsigned a, unsigned b) {
  unsigned d;
  asm("v_pk_mul_bf16 %0, %1, %2" : "=v"(d) : "v"(a), "v"(b));
  return d;
}

__device__ __forceinline__ uint4 pack8(float4 a, float4 b) {
  union { uint4 u; __bf16 h[8]; } r;
  r.h[0] = (__bf16)a.x; r.h[1] = (__bf16)a.y;
  r.h[2] = (__bf16)a.z; r.h[3] = (__bf16)a.w;
  r.h[4] = (__bf16)b.x; r.h[5] = (__bf16)b.y;
  r.h[6] = (__bf16)b.z; r.h[7] = (__bf16)b.w;
  return r.u;
}

__device__ __forceinline__ v8f wmma_bf16(const Frag& a, const Frag& b, v8f c) {
  return __builtin_amdgcn_wmma_f32_16x16x32_bf16(false, a.v, false, b.v,
                                                 (short)0, c, false, false);
}

// ---- prologue: W f32 -> bf16 (packed) into workspace, once per launch ----
__global__ __launch_bounds__(256)
void convert_w_bf16(const float* __restrict__ W, uint4* __restrict__ Wq) {
  const size_t i = (size_t)blockIdx.x * blockDim.x + threadIdx.x; // one uint4 each
  const float* p = W + i * 8;
  F4 a, b;
  a.v = *(const float4*)p;
  b.v = *(const float4*)(p + 4);
  Wq[i] = pack8(a.v, b.v);
}

template <bool PRE>
__global__ __launch_bounds__(256)
void cin_bilinear_wmma(const float* __restrict__ x0,
                       const float* __restrict__ x,
                       const float* __restrict__ W,     // f32 (fallback path)
                       const uint4* __restrict__ Wq,    // bf16-packed (fast path)
                       const float* __restrict__ bias,
                       float* __restrict__ out)
{
  __shared__ uint4 sX[BM * XROW];        // x tile, bf16, 34816 B
  __shared__ uint4 sW[FB * BN * WROW];   // W slice, bf16, 20480 B

  const int tid  = threadIdx.x;
  const int lane = tid & 31;
  const int wid  = tid >> 5;
  const int waveM = wid & 3;   // 0..3 -> 32-row band
  const int waveN = wid >> 2;  // 0..1 -> 32-col band
  const int b0 = blockIdx.x * BM;
  const int o0 = blockIdx.y * BN;
  const int l16 = lane & 15;
  const bool hi = lane >= 16;

  // ---- stage x[b0..b0+127][0..127] as bf16 into sX ----
  {
    const int r = tid >> 1, h = tid & 1;
    const float* xp = x + (size_t)(b0 + r) * HPREV + h * 64;
    #pragma unroll
    for (int i = 0; i < 8; ++i) {
      F4 a, b;
      a.v = *(const float4*)(xp + i * 8);
      b.v = *(const float4*)(xp + i * 8 + 4);
      sX[r * XROW + h * 8 + i] = pack8(a.v, b.v);
    }
  }
  __syncthreads();

  v8f acc[2][2];
  #pragma unroll
  for (int i = 0; i < 2; ++i)
    #pragma unroll
    for (int j = 0; j < 2; ++j)
      #pragma unroll
      for (int k = 0; k < 8; ++k) acc[i][j][k] = 0.0f;

  // A-frag (16-bit 16x32): lanes 0-15 row M=l16 hold K {0..7,16..23};
  // lanes 16-31 same rows hold K {8..15,24..31}.
  const int rowL0 = waveM * 32 + l16;   // mt = 0
  const int rowL1 = rowL0 + 16;         // mt = 1
  const int aSel  = hi ? 1 : 0;
  // B-frag (32x16): lane = column N, K {0..7,16..23} / {8..15,24..31}
  const int nL0   = waveN * 32 + l16;
  const int bSel  = hi ? 1 : 0;

  // W staging assignment: thread -> (n, quarter-of-32-k)
  const int stageN = tid >> 2;  // 0..63
  const int stageQ = tid & 3;   // 0..3
  const float* WbaseF = W + (size_t)(o0 + stageN) * KTOT + stageQ * 8;
  const uint4* WbaseQ = Wq + (size_t)(o0 + stageN) * (KTOT / 8) + stageQ;

  for (int pc = 0; pc < 4; ++pc) {          // p-chunk of 32
    // x-fragments for this p-chunk; reused across all 128 f values
    Frag xf0, xf1;
    {
      const int i0 = rowL0 * XROW + pc * 4 + aSel;
      xf0.u2[0] = sX[i0]; xf0.u2[1] = sX[i0 + 2];
      const int i1 = rowL1 * XROW + pc * 4 + aSel;
      xf1.u2[0] = sX[i1]; xf1.u2[1] = sX[i1 + 2];
    }

    // parity-double-buffered fragment registers (avoid WMMA WAR nop bubbles)
    Frag aBuf[2][2], bBuf[2][2];

    for (int fb = 0; fb < FIELD / FB; ++fb) {
      const int f0 = fb * FB;

      // x0 scalars for this super-step: issue early so latency hides under staging
      F4 q0, q1;
      q0.v = *(const float4*)(x0 + (size_t)(b0 + rowL0) * FIELD + f0);
      q1.v = *(const float4*)(x0 + (size_t)(b0 + rowL1) * FIELD + f0);

      __syncthreads();   // previous sW consumers done
      // stage W[o0..o0+63][f0..f0+3][pc*32..pc*32+31] -> bf16 LDS
      if (PRE) {
        #pragma unroll
        for (int ff = 0; ff < FB; ++ff) {
          const uint4 v = WbaseQ[(size_t)(f0 + ff) * (HPREV / 8) + pc * 4];
          sW[(ff * BN + stageN) * WROW + stageQ] = v;
        }
      } else {
        #pragma unroll
        for (int ff = 0; ff < FB; ++ff) {
          const float* wp = WbaseF + (size_t)(f0 + ff) * HPREV + pc * 32;
          F4 a, b;
          a.v = *(const float4*)(wp);
          b.v = *(const float4*)(wp + 4);
          sW[(ff * BN + stageN) * WROW + stageQ] = pack8(a.v, b.v);
        }
      }
      __syncthreads();

      #pragma unroll
      for (int ff = 0; ff < FB; ++ff) {
        const unsigned s0 = splat_bf16(q0.f[ff]);
        const unsigned s1 = splat_bf16(q1.f[ff]);
        Frag& a0 = aBuf[ff & 1][0];
        Frag& a1 = aBuf[ff & 1][1];
        #pragma unroll
        for (int i = 0; i < 8; ++i) {
          a0.w[i] = pk_mul_bf16(xf0.w[i], s0);
          a1.w[i] = pk_mul_bf16(xf1.w[i], s1);
        }
        Frag& bf0 = bBuf[ff & 1][0];
        Frag& bf1 = bBuf[ff & 1][1];
        const uint4* w0 = &sW[(ff * BN + nL0) * WROW + bSel];
        bf0.u2[0] = w0[0]; bf0.u2[1] = w0[2];
        const uint4* w1 = w0 + 16 * WROW;
        bf1.u2[0] = w1[0]; bf1.u2[1] = w1[2];

        acc[0][0] = wmma_bf16(a0, bf0, acc[0][0]);
        acc[0][1] = wmma_bf16(a0, bf1, acc[0][1]);
        acc[1][0] = wmma_bf16(a1, bf0, acc[1][0]);
        acc[1][1] = wmma_bf16(a1, bf1, acc[1][1]);
      }
    }
  }

  // ---- epilogue: C layout VGPR r -> M = r (+8 for lanes>=16), N = l16 ----
  #pragma unroll
  for (int nt = 0; nt < 2; ++nt) {
    const int colG = o0 + waveN * 32 + nt * 16 + l16;
    const float bb = bias[colG];
    #pragma unroll
    for (int mt = 0; mt < 2; ++mt) {
      const int rowG = b0 + waveM * 32 + mt * 16 + (hi ? 8 : 0);
      #pragma unroll
      for (int r = 0; r < 8; ++r) {
        __builtin_nontemporal_store(acc[mt][nt][r] + bb,
                                    out + (size_t)(rowG + r) * HNEXT + colG);
      }
    }
  }
}

extern "C" void kernel_launch(void* const* d_in, const int* in_sizes, int n_in,
                              void* d_out, int out_size, void* d_ws, size_t ws_size,
                              hipStream_t stream) {
  (void)in_sizes; (void)n_in; (void)out_size;
  const float* x0 = (const float*)d_in[0];   // [4096,128]
  const float* x  = (const float*)d_in[1];   // [4096,128]
  const float* W  = (const float*)d_in[2];   // [256,128,128]
  const float* b  = (const float*)d_in[3];   // [256]
  float* out = (float*)d_out;                // [4096,256]

  dim3 grid(B_TOT / BM, HNEXT / BN);         // 32 x 4
  dim3 block(256);                           // 8 waves (wave32)

  const size_t wq_bytes = (size_t)HNEXT * KTOT * sizeof(unsigned short); // 8.39 MB
  if (ws_size >= wq_bytes) {
    uint4* Wq = (uint4*)d_ws;
    // 256*16384 floats / 8 per thread = 524288 threads
    convert_w_bf16<<<dim3((HNEXT * KTOT / 8) / 256), dim3(256), 0, stream>>>(W, Wq);
    cin_bilinear_wmma<true><<<grid, block, 0, stream>>>(x0, x, W, Wq, b, out);
  } else {
    cin_bilinear_wmma<false><<<grid, block, 0, stream>>>(x0, x, W, nullptr, b, out);
  }
}
